// FuzzyPVConv_16681652977961
// MI455X (gfx1250) — matrix operand order repaired and checked by reading, not compiled, not run on hardware
//
#include <hip/hip_runtime.h>
#include <math.h>

// ---------------------------------------------------------------------------
// FuzzyPVConv for MI455X (gfx1250, wave32, WMMA bf16 16x16x32 / f32 acc)
//   - activation voxel grid: channel-last [B, R3, 64] bf16
//   - conv reduction order:  k' = (kd*9+kh*3+kw)*64 + ci  (tap-major)
//   - 1x1-conv GEMM inputs:  channel-last [B, N, K] bf16
//   - conv wave tile: 64 outch x 32 voxels (full Z row) -> 2x A reuse
// ---------------------------------------------------------------------------

#define BB   4
#define CH   64
#define NP   4096
#define RR   32
#define R3   32768           // 32^3
#define CONVK 1728           // 27 * 64
#define CONVCHUNKS 54        // 1728 / 32

typedef __attribute__((ext_vector_type(16))) __bf16 v16bf;
typedef __attribute__((ext_vector_type(8)))  float  v8f;

__device__ __forceinline__ v8f wmma_bf16(v16bf a, v16bf b, v8f c) {
  return __builtin_amdgcn_wmma_f32_16x16x32_bf16(false, a, false, b,
                                                 (short)0, c, false, false);
}

// A-operand k index inside a 32-wide chunk for (lane, element e of v16bf)
__device__ __forceinline__ int a_kk(int lane, int e) {
  int hi = lane >> 4;
  return hi * 16 + e;
}

// ---------------------------------------------------------------------------
// 1) Per-batch coord prep: voxel coords nc in [0,R-1] and unit dirs cn
// ---------------------------------------------------------------------------
__global__ void k_coord_prep(const float* __restrict__ coords,
                             float* __restrict__ nc, float* __restrict__ cn) {
  int b = blockIdx.x, tid = threadIdx.x;
  __shared__ float sm[256];
  __shared__ float mean[3];
  __shared__ float mnorm;
  const float* cb = coords + (size_t)b * 3 * NP;
  for (int d = 0; d < 3; ++d) {
    float s = 0.f;
    for (int n = tid; n < NP; n += 256) s += cb[d * NP + n];
    sm[tid] = s; __syncthreads();
    for (int o = 128; o > 0; o >>= 1) {
      if (tid < o) sm[tid] += sm[tid + o];
      __syncthreads();
    }
    if (tid == 0) mean[d] = sm[0] * (1.0f / NP);
    __syncthreads();
  }
  float mx = 0.f;
  for (int n = tid; n < NP; n += 256) {
    float x = cb[n] - mean[0], y = cb[NP + n] - mean[1], z = cb[2 * NP + n] - mean[2];
    mx = fmaxf(mx, sqrtf(x * x + y * y + z * z));
  }
  sm[tid] = mx; __syncthreads();
  for (int o = 128; o > 0; o >>= 1) {
    if (tid < o) sm[tid] = fmaxf(sm[tid], sm[tid + o]);
    __syncthreads();
  }
  if (tid == 0) mnorm = sm[0];
  __syncthreads();
  float inv2 = 0.5f / mnorm;
  float* ncb = nc + (size_t)b * 3 * NP;
  float* cnb = cn + (size_t)b * 3 * NP;
  for (int n = tid; n < NP; n += 256) {
    #pragma unroll
    for (int d = 0; d < 3; ++d) {
      float v = (cb[d * NP + n] - mean[d]) * inv2 + 0.5f;
      v = fminf(fmaxf(v * (float)RR, 0.f), (float)(RR - 1));
      ncb[d * NP + n] = v;
    }
    float x = cb[n], y = cb[NP + n], z = cb[2 * NP + n];
    float ir = 1.f / fmaxf(sqrtf(x * x + y * y + z * z), 1e-12f);
    cnb[n] = x * ir; cnb[NP + n] = y * ir; cnb[2 * NP + n] = z * ir;
  }
}

// ---------------------------------------------------------------------------
// 2) Scatter-mean voxelization; 4 threads per point (16 channels each)
// ---------------------------------------------------------------------------
__global__ void k_scatter(const float* __restrict__ feat,
                          const float* __restrict__ nc,
                          float* __restrict__ gsum, float* __restrict__ cnt) {
  int idx = blockIdx.x * 256 + threadIdx.x;   // B * N * 4
  if (idx >= BB * NP * 4) return;
  int n = idx & (NP - 1);
  int b = (idx >> 12) & 3;
  int cg = idx >> 14;
  const float* nb = nc + (size_t)b * 3 * NP;
  int vx = (int)rintf(nb[n]);
  int vy = (int)rintf(nb[NP + n]);
  int vz = (int)rintf(nb[2 * NP + n]);
  int flat = (vx * RR + vy) * RR + vz;
  if (cg == 0) atomicAdd(&cnt[(size_t)b * R3 + flat], 1.0f);
  const float* fb = feat + (size_t)b * CH * NP;
  float* gb = gsum + (size_t)b * CH * R3;
  for (int c = cg * 16; c < cg * 16 + 16; ++c)
    atomicAdd(&gb[((size_t)c << 15) + flat], fb[c * NP + n]);
}

// divide by counts, emit channel-last bf16 grid [B, R3, 64]
__global__ void k_grid_avg(const float* __restrict__ gsum,
                           const float* __restrict__ cnt,
                           __bf16* __restrict__ gbf) {
  int idx = blockIdx.x * 256 + threadIdx.x;   // over B*R3
  if (idx >= BB * R3) return;
  int b = idx >> 15, v = idx & (R3 - 1);
  float ic = 1.f / fmaxf(cnt[idx], 1.f);
  const float* gb = gsum + ((size_t)b << 21);
  __bf16* ob = gbf + ((size_t)idx << 6);
  for (int c = 0; c < CH; ++c)
    ob[c] = (__bf16)(gb[((size_t)c << 15) + v] * ic);
}

// ---------------------------------------------------------------------------
// 3) Weight swizzlers -> A-operand layout bf16 [4][K/32][32][16]
// ---------------------------------------------------------------------------
__global__ void k_wswz(const float* __restrict__ W, int K, __bf16* __restrict__ dst) {
  int idx = blockIdx.x * 256 + threadIdx.x;
  if (idx >= 64 * K) return;
  int nch = K >> 5;
  int e = idx & 15;
  int lane = (idx >> 4) & 31;
  int te = idx >> 9;
  int t = te % nch;
  int g = te / nch;
  int o = g * 16 + (lane & 15);
  int k = t * 32 + a_kk(lane, e);
  dst[idx] = (__bf16)W[(size_t)o * K + k];
}

// conv: W[64, 64, 27] (o, ci, kidx) with k' = kidx*64 + ci
__global__ void k_wswz_conv(const float* __restrict__ W, __bf16* __restrict__ dst) {
  int idx = blockIdx.x * 256 + threadIdx.x;
  if (idx >= CH * CONVK) return;
  int e = idx & 15;
  int lane = (idx >> 4) & 31;
  int te = idx >> 9;
  int t = te % CONVCHUNKS;
  int g = te / CONVCHUNKS;
  int o = g * 16 + (lane & 15);
  int k = t * 32 + a_kk(lane, e);
  int ci = k & 63, kidx = k >> 6;
  dst[idx] = (__bf16)W[(size_t)o * CONVK + ci * 27 + kidx];
}

// ---------------------------------------------------------------------------
// 4) 3x3x3 conv as implicit GEMM.  Wave = 64 outch x 32 voxels (full Z row).
//    Per tap: 1 boundary test per column-tile, contiguous v16bf B loads,
//    4 A tiles reused across both column tiles -> 8 WMMAs / chunk.
// ---------------------------------------------------------------------------
__global__ void __launch_bounds__(256)
k_conv3d_wmma(const __bf16* __restrict__ xin,   // [B,R3,64] bf16 channel-last
              const __bf16* __restrict__ wswz,  // [4][54][32][16], k'=kidx*64+ci
              float* __restrict__ out) {        // [B,64,R3] f32
  int wv = (blockIdx.x << 3) + (threadIdx.x >> 5);   // 4096 waves
  int lane = threadIdx.x & 31;
  int b = wv >> 10;              // 1024 tiles per batch
  int tt = wv & 1023;
  int Y = tt & 31;
  int X = tt >> 5;
  int col = lane & 15;
  int hi = lane >> 4;

  v8f aA0 = {}, aA1 = {}, aA2 = {}, aA3 = {};
  v8f aB0 = {}, aB1 = {}, aB2 = {}, aB3 = {};
  const __bf16* xb = xin + ((size_t)b << 21);   // b * R3 * 64
  const size_t gs = (size_t)CONVCHUNKS * 512;   // per out-group stride in wswz

  int t = 0;
  for (int kd = 0; kd < 3; ++kd)
    for (int kh = 0; kh < 3; ++kh) {
      int xx = X + kd - 1, yy = Y + kh - 1;
      bool inxy = ((unsigned)xx < 32u) & ((unsigned)yy < 32u);
      const __bf16* rowp =
          xb + ((((long)xx << 10) + ((long)yy << 5)) << 6) + hi * 16;
      for (int kw = 0; kw < 3; ++kw) {
        int zza = col + kw - 1;         // column tile A: Z in [0,16)
        int zzb = zza + 16;             // column tile B: Z in [16,32)
        bool inba = inxy & ((unsigned)zza < 32u);
        bool inbb = inxy & (zzb < 32);
        const __bf16* pA = rowp + ((long)zza << 6);
        const __bf16* pB = rowp + ((long)zzb << 6);
        #pragma unroll
        for (int half = 0; half < 2; ++half, ++t) {
          v16bf vba = {}, vbb = {};
          if (inba) vba = *(const v16bf*)(pA + half * 32);
          if (inbb) vbb = *(const v16bf*)(pB + half * 32);
          const __bf16* wt = wswz + ((size_t)t * 32 + lane) * 16;
          v16bf a0 = *(const v16bf*)(wt);
          v16bf a1 = *(const v16bf*)(wt + gs);
          v16bf a2 = *(const v16bf*)(wt + 2 * gs);
          v16bf a3 = *(const v16bf*)(wt + 3 * gs);
          aA0 = wmma_bf16(a0, vba, aA0);
          aB0 = wmma_bf16(a0, vbb, aB0);
          aA1 = wmma_bf16(a1, vba, aA1);
          aB1 = wmma_bf16(a1, vbb, aB1);
          aA2 = wmma_bf16(a2, vba, aA2);
          aB2 = wmma_bf16(a2, vbb, aB2);
          aA3 = wmma_bf16(a3, vba, aA3);
          aB3 = wmma_bf16(a3, vbb, aB3);
        }
      }
    }

  size_t voxA = ((size_t)X << 10) + (Y << 5) + col;
  size_t voxB = voxA + 16;
  float* ob = out + ((size_t)b << 21);
  #pragma unroll
  for (int r = 0; r < 8; ++r) {
    int o0 = r + 8 * hi;
    ob[((size_t)o0 << 15) + voxA] = aA0[r];
    ob[((size_t)o0 << 15) + voxB] = aB0[r];
    ob[((size_t)(o0 + 16) << 15) + voxA] = aA1[r];
    ob[((size_t)(o0 + 16) << 15) + voxB] = aB1[r];
    ob[((size_t)(o0 + 32) << 15) + voxA] = aA2[r];
    ob[((size_t)(o0 + 32) << 15) + voxB] = aB2[r];
    ob[((size_t)(o0 + 48) << 15) + voxA] = aA3[r];
    ob[((size_t)(o0 + 48) << 15) + voxB] = aB3[r];
  }
}

// ---------------------------------------------------------------------------
// 5) Per-channel BN statistics over (B, S): biased var, eps 1e-4
// ---------------------------------------------------------------------------
__global__ void k_chan_stats(const float* __restrict__ x, int S,
                             float* __restrict__ mean, float* __restrict__ invstd) {
  int c = blockIdx.x, tid = threadIdx.x;
  __shared__ float s1[256], s2[256];
  float a = 0.f, q = 0.f;
  for (int b = 0; b < BB; ++b) {
    const float* xb = x + ((size_t)(b * CH + c)) * S;
    for (int i = tid; i < S; i += 256) { float v = xb[i]; a += v; q += v * v; }
  }
  s1[tid] = a; s2[tid] = q; __syncthreads();
  for (int o = 128; o > 0; o >>= 1) {
    if (tid < o) { s1[tid] += s1[tid + o]; s2[tid] += s2[tid + o]; }
    __syncthreads();
  }
  if (tid == 0) {
    float cnt = (float)BB * (float)S;
    float m = s1[0] / cnt;
    float v = s2[0] / cnt - m * m;
    mean[c] = m;
    invstd[c] = rsqrtf(v + 1e-4f);
  }
}

// BN + (leaky)ReLU.  x is [B,64,S] channel-major f32.
// optional outF: same layout f32;  optional outBF: channel-last [B,S,CT] at coff
__global__ void k_bn_act(const float* __restrict__ x,
                         const float* __restrict__ mean, const float* __restrict__ invstd,
                         const float* __restrict__ g, const float* __restrict__ beta,
                         float slope, int S,
                         float* __restrict__ outF,
                         __bf16* __restrict__ outBF, int CT, int coff) {
  size_t idx = (size_t)blockIdx.x * 256 + threadIdx.x;
  size_t total = (size_t)BB * CH * S;
  if (idx >= total) return;
  size_t s = idx % (size_t)S;
  int c = (int)((idx / S) & 63);
  int b = (int)(idx / ((size_t)CH * S));
  float v = (x[idx] - mean[c]) * invstd[c] * g[c] + beta[c];
  v = v > 0.f ? v : slope * v;
  if (outF) outF[idx] = v;
  if (outBF) outBF[((size_t)b * S + s) * CT + coff + c] = (__bf16)v;
}

// ---------------------------------------------------------------------------
// 6) Trilinear devoxelization -> bf16 into comb[:, :, 0:64] (channel-last)
// ---------------------------------------------------------------------------
__global__ void k_devox(const float* __restrict__ v, const float* __restrict__ nc,
                        __bf16* __restrict__ comb) {
  int idx = blockIdx.x * 256 + threadIdx.x;
  if (idx >= BB * NP) return;
  int b = idx >> 12, n = idx & (NP - 1);
  const float* nb = nc + (size_t)b * 3 * NP;
  float fx = nb[n], fy = nb[NP + n], fz = nb[2 * NP + n];
  int lx = (int)floorf(fx), ly = (int)floorf(fy), lz = (int)floorf(fz);
  float rx = fx - lx, ry = fy - ly, rz = fz - lz;
  int hx = lx + 1 > 31 ? 31 : lx + 1;
  int hy = ly + 1 > 31 ? 31 : ly + 1;
  int hz = lz + 1 > 31 ? 31 : lz + 1;
  int fl[8]; float w[8]; int j = 0;
  #pragma unroll
  for (int dx = 0; dx < 2; ++dx)
    #pragma unroll
    for (int dy = 0; dy < 2; ++dy)
      #pragma unroll
      for (int dz = 0; dz < 2; ++dz) {
        int ix = dx ? hx : lx, iy = dy ? hy : ly, iz = dz ? hz : lz;
        fl[j] = (ix * RR + iy) * RR + iz;
        w[j] = (dx ? rx : 1.f - rx) * (dy ? ry : 1.f - ry) * (dz ? rz : 1.f - rz);
        ++j;
      }
  const float* vb = v + ((size_t)b << 21);
  __bf16* ob = comb + (size_t)idx * 192;
  for (int c = 0; c < CH; ++c) {
    const float* vc = vb + ((size_t)c << 15);
    float acc = 0.f;
    #pragma unroll
    for (int k = 0; k < 8; ++k) acc += w[k] * vc[fl[k]];
    ob[c] = (__bf16)acc;
  }
}

// ---------------------------------------------------------------------------
// 7) Attention pass 1: wave per query; lanes stride over keys, shuffle-merge
//    (max, sumexp) pairs.
// ---------------------------------------------------------------------------
__global__ void __launch_bounds__(256)
k_attn_stats(const float* __restrict__ cn,
             float* __restrict__ rmax, float* __restrict__ rsum) {
  int wv = (blockIdx.x << 3) + (threadIdx.x >> 5);  // B*N waves
  int lane = threadIdx.x & 31;
  int b = wv >> 12, n = wv & (NP - 1);
  const float* cb = cn + (size_t)b * 3 * NP;
  float qx = cb[n], qy = cb[NP + n], qz = cb[2 * NP + n];
  float mx = -1e30f, sm = 0.f;
  for (int m = lane; m < NP; m += 32) {
    float s = 10.f * (qx * cb[m] + qy * cb[NP + m] + qz * cb[2 * NP + m]);
    if (s > mx) { sm = sm * __expf(mx - s) + 1.f; mx = s; }
    else        { sm += __expf(s - mx); }
  }
  #pragma unroll
  for (int o = 16; o > 0; o >>= 1) {
    float mo = __shfl_xor(mx, o, 32);
    float so = __shfl_xor(sm, o, 32);
    float nm = fmaxf(mx, mo);
    sm = sm * __expf(mx - nm) + so * __expf(mo - nm);
    mx = nm;
  }
  if (lane == 0) { rmax[wv] = mx; rsum[wv] = sm; }
}

// ---------------------------------------------------------------------------
// 8) Attention pass 2: fz = P @ V with WMMA.  Wave = 16 queries x 64 channels.
//    Keys loaded once per lane per chunk; A-layout permutation via __shfl.
//    V tiles contiguous v16bf; output channel-last [B, N, 64].
// ---------------------------------------------------------------------------
__global__ void __launch_bounds__(256)
k_attn_av(const float* __restrict__ cn,
          const float* __restrict__ rmax, const float* __restrict__ rsum,
          const __bf16* __restrict__ featbf,   // [B,64,4096]
          __bf16* __restrict__ fzattn) {       // [B,4096,64]
  int wv = (blockIdx.x << 3) + (threadIdx.x >> 5);
  int lane = threadIdx.x & 31;
  int b = wv >> 8;
  int qbase = (wv & 255) << 4;
  int col = lane & 15, hi = lane >> 4;
  int q = qbase + col;                 // A rows: M = lane % 16
  const float* cb = cn + (size_t)b * 3 * NP;
  float qx = cb[q], qy = cb[NP + q], qz = cb[2 * NP + q];
  float rm = rmax[(b << 12) + q];
  float ri = 1.f / rsum[(b << 12) + q];
  const __bf16* fb = featbf + ((size_t)b << 18);
  v8f acc0 = {}, acc1 = {}, acc2 = {}, acc3 = {};
  for (int t = 0; t < 128; ++t) {
    int mb = t << 5;
    float kxl = cb[mb + lane];
    float kyl = cb[NP + mb + lane];
    float kzl = cb[2 * NP + mb + lane];
    v16bf pa;
    #pragma unroll
    for (int e = 0; e < 16; ++e) {
      int src = a_kk(lane, e);
      float kx = __shfl(kxl, src, 32);
      float ky = __shfl(kyl, src, 32);
      float kz = __shfl(kzl, src, 32);
      float s = 10.f * (qx * kx + qy * ky + qz * kz);
      pa[e] = (__bf16)(__expf(s - rm) * ri);
    }
    int kb = mb + hi * 16;
    v16bf v0 = *(const v16bf*)(fb + ((size_t)col << 12) + kb);
    v16bf v1 = *(const v16bf*)(fb + ((size_t)(16 + col) << 12) + kb);
    v16bf v2 = *(const v16bf*)(fb + ((size_t)(32 + col) << 12) + kb);
    v16bf v3 = *(const v16bf*)(fb + ((size_t)(48 + col) << 12) + kb);
    acc0 = wmma_bf16(pa, v0, acc0);
    acc1 = wmma_bf16(pa, v1, acc1);
    acc2 = wmma_bf16(pa, v2, acc2);
    acc3 = wmma_bf16(pa, v3, acc3);
  }
  // D: row = query (r + 8*hi), col = channel (lane%16); store channel-last
  #pragma unroll
  for (int r = 0; r < 8; ++r) {
    int qq = qbase + r + 8 * hi;
    __bf16* orow = fzattn + (((size_t)(b << 12) + qq) << 6);
    orow[col] = (__bf16)acc0[r];
    orow[16 + col] = (__bf16)acc1[r];
    orow[32 + col] = (__bf16)acc2[r];
    orow[48 + col] = (__bf16)acc3[r];
  }
}

// ---------------------------------------------------------------------------
// 9) Generic 1x1-conv GEMM: Y[b,o,n] = sum_k W[o,k]*X[b,n,k] (+bias[o]).
// ---------------------------------------------------------------------------
__global__ void __launch_bounds__(256)
k_gemm_wmma(const __bf16* __restrict__ wswz,
            const float* __restrict__ bias,
            const __bf16* __restrict__ X, int K,     // [B, 4096, K]
            float* __restrict__ outF,                // [B, 64, 4096]
            __bf16* __restrict__ outBF, int CT, int coff) { // [B,4096,CT]
  int wv = (blockIdx.x << 3) + (threadIdx.x >> 5);
  int lane = threadIdx.x & 31;
  int b = wv >> 8;
  int nbase = (wv & 255) << 4;
  int col = lane & 15, hi = lane >> 4;
  int n = nbase + col;
  int nch = K >> 5;
  const size_t gs = (size_t)nch * 512;
  const __bf16* xr = X + ((size_t)(b * NP + n)) * K + hi * 16;
  v8f acc0 = {}, acc1 = {}, acc2 = {}, acc3 = {};
  for (int t = 0; t < nch; ++t) {
    __builtin_prefetch(xr + (t + 1) * 32, 0, 0);   // global_prefetch next chunk
    v16bf vb = *(const v16bf*)(xr + t * 32);
    const __bf16* wt = wswz + ((size_t)t * 32 + lane) * 16;
    v16bf a0 = *(const v16bf*)(wt);
    v16bf a1 = *(const v16bf*)(wt + gs);
    v16bf a2 = *(const v16bf*)(wt + 2 * gs);
    v16bf a3 = *(const v16bf*)(wt + 3 * gs);
    acc0 = wmma_bf16(a0, vb, acc0);
    acc1 = wmma_bf16(a1, vb, acc1);
    acc2 = wmma_bf16(a2, vb, acc2);
    acc3 = wmma_bf16(a3, vb, acc3);
  }
  #pragma unroll
  for (int r = 0; r < 8; ++r) {
    int o0 = r + 8 * hi;
    #pragma unroll
    for (int g = 0; g < 4; ++g) {
      int o = g * 16 + o0;
      float v = (g == 0 ? acc0[r] : g == 1 ? acc1[r] : g == 2 ? acc2[r] : acc3[r]);
      if (bias) v += bias[o];
      if (outF) outF[((size_t)(b * CH + o)) * NP + n] = v;
      if (outBF) outBF[((size_t)(b * NP + n)) * CT + coff + o] = (__bf16)v;
    }
  }
}

// f32 [B,64,N] -> bf16 same layout (attention V)
__global__ void k_f2bf(const float* __restrict__ x, __bf16* __restrict__ y, int total) {
  int i = blockIdx.x * 256 + threadIdx.x;
  if (i < total) y[i] = (__bf16)x[i];
}

// f32 [B,64,N] -> bf16 channel-last [B,N,64] (point-GEMM input)
__global__ void k_feat_cl(const float* __restrict__ x, __bf16* __restrict__ y) {
  int idx = blockIdx.x * 256 + threadIdx.x;
  if (idx >= BB * NP) return;
  int b = idx >> 12, n = idx & (NP - 1);
  const float* fb = x + ((size_t)b << 18);
  __bf16* ob = y + ((size_t)idx << 6);
  for (int c = 0; c < CH; ++c) ob[c] = (__bf16)fb[((size_t)c << 12) + n];
}

// ---------------------------------------------------------------------------
extern "C" void kernel_launch(void* const* d_in, const int* in_sizes, int n_in,
                              void* d_out, int out_size, void* d_ws, size_t ws_size,
                              hipStream_t stream) {
  (void)in_sizes; (void)n_in; (void)out_size; (void)ws_size;

  const float* features = (const float*)d_in[0];
  const float* coords   = (const float*)d_in[1];
  const float* w1  = (const float*)d_in[2];
  const float* g1  = (const float*)d_in[4];
  const float* be1 = (const float*)d_in[5];
  const float* w2  = (const float*)d_in[6];
  const float* g2  = (const float*)d_in[8];
  const float* be2 = (const float*)d_in[9];
  const float* wp  = (const float*)d_in[10];
  const float* gp  = (const float*)d_in[12];
  const float* bep = (const float*)d_in[13];
  const float* wf  = (const float*)d_in[14];
  const float* bf  = (const float*)d_in[15];
  const float* wfu = (const float*)d_in[16];
  const float* gfu = (const float*)d_in[18];
  const float* befu = (const float*)d_in[19];
  float* outp = (float*)d_out;

  // ---- workspace carve-out ----
  size_t off = 0;
  char* base = (char*)d_ws;
  auto carve = [&](size_t bytes) -> void* {
    off = (off + 255) & ~(size_t)255;
    void* p = base + off;
    off += bytes;
    return p;
  };
  float*  nc      = (float*)carve((size_t)BB * 3 * NP * 4);
  float*  cn      = (float*)carve((size_t)BB * 3 * NP * 4);
  float*  rmax    = (float*)carve((size_t)BB * NP * 4);
  float*  rsum    = (float*)carve((size_t)BB * NP * 4);
  float*  cnt     = (float*)carve((size_t)BB * R3 * 4);
  float*  vf32    = (float*)carve((size_t)BB * CH * R3 * 4);   // gsum / conv outs
  __bf16* gridbf  = (__bf16*)carve((size_t)BB * R3 * CH * 2);  // channel-last acts
  __bf16* featbf  = (__bf16*)carve((size_t)BB * CH * NP * 2);  // channel-major
  __bf16* featcl  = (__bf16*)carve((size_t)BB * NP * CH * 2);  // channel-last
  __bf16* fzattn  = (__bf16*)carve((size_t)BB * NP * CH * 2);  // channel-last
  float*  ptlin   = (float*)carve((size_t)BB * CH * NP * 4);
  float*  fuslin  = (float*)carve((size_t)BB * CH * NP * 4);
  __bf16* comb    = (__bf16*)carve((size_t)BB * NP * 192 * 2); // channel-last
  __bf16* w1swz   = (__bf16*)carve((size_t)CH * CONVK * 2);
  __bf16* w2swz   = (__bf16*)carve((size_t)CH * CONVK * 2);
  __bf16* wpswz   = (__bf16*)carve((size_t)CH * CH * 2);
  __bf16* wfswz   = (__bf16*)carve((size_t)CH * CH * 2);
  __bf16* wfuswz  = (__bf16*)carve((size_t)CH * 192 * 2);
  float*  stats   = (float*)carve(512 * 4);
  float *m1 = stats, *s1 = stats + 64, *m2 = stats + 128, *s2 = stats + 192;
  float *mp = stats + 256, *sp = stats + 320, *mfu = stats + 384, *sfu = stats + 448;

  // ---- prep ----
  k_coord_prep<<<BB, 256, 0, stream>>>(coords, nc, cn);
  k_f2bf<<<(BB * CH * NP) / 256, 256, 0, stream>>>(features, featbf, BB * CH * NP);
  k_feat_cl<<<(BB * NP) / 256, 256, 0, stream>>>(features, featcl);
  k_wswz_conv<<<(CH * CONVK + 255) / 256, 256, 0, stream>>>(w1, w1swz);
  k_wswz_conv<<<(CH * CONVK + 255) / 256, 256, 0, stream>>>(w2, w2swz);
  k_wswz<<<(CH * CH + 255) / 256, 256, 0, stream>>>(wp, CH, wpswz);
  k_wswz<<<(CH * CH + 255) / 256, 256, 0, stream>>>(wf, CH, wfswz);
  k_wswz<<<(CH * 192 + 255) / 256, 256, 0, stream>>>(wfu, 192, wfuswz);

  // ---- voxel branch ----
  hipMemsetAsync(vf32, 0, (size_t)BB * CH * R3 * 4, stream);
  hipMemsetAsync(cnt, 0, (size_t)BB * R3 * 4, stream);
  k_scatter<<<(BB * NP * 4) / 256, 256, 0, stream>>>(features, nc, vf32, cnt);
  k_grid_avg<<<(BB * R3) / 256, 256, 0, stream>>>(vf32, cnt, gridbf);

  // conv1 -> BN(leaky 0.1) -> channel-last bf16 act (overwrites gridbf)
  k_conv3d_wmma<<<512, 256, 0, stream>>>(gridbf, w1swz, vf32);
  k_chan_stats<<<CH, 256, 0, stream>>>(vf32, R3, m1, s1);
  k_bn_act<<<(BB * CH * R3) / 256, 256, 0, stream>>>(vf32, m1, s1, g1, be1, 0.1f,
                                                     R3, nullptr, gridbf, CH, 0);
  // conv2 -> BN(leaky 0.1) in place (f32 for devox)
  k_conv3d_wmma<<<512, 256, 0, stream>>>(gridbf, w2swz, vf32);
  k_chan_stats<<<CH, 256, 0, stream>>>(vf32, R3, m2, s2);
  k_bn_act<<<(BB * CH * R3) / 256, 256, 0, stream>>>(vf32, m2, s2, g2, be2, 0.1f,
                                                     R3, vf32, nullptr, 0, 0);
  k_devox<<<(BB * NP) / 256, 256, 0, stream>>>(vf32, nc, comb);  // comb[:,:,0:64]

  // ---- point branch ----  (bias bp cancels under BN)
  k_gemm_wmma<<<128, 256, 0, stream>>>(wpswz, nullptr, featcl, CH,
                                       ptlin, nullptr, 0, 0);
  k_chan_stats<<<CH, 256, 0, stream>>>(ptlin, NP, mp, sp);
  k_bn_act<<<(BB * CH * NP) / 256, 256, 0, stream>>>(ptlin, mp, sp, gp, bep, 0.0f,
                                                     NP, nullptr, comb, 192, CH);

  // ---- fuzzy branch ----
  k_attn_stats<<<(BB * NP) / 8, 256, 0, stream>>>(cn, rmax, rsum);
  k_attn_av<<<128, 256, 0, stream>>>(cn, rmax, rsum, featbf, fzattn);
  k_gemm_wmma<<<128, 256, 0, stream>>>(wfswz, bf, fzattn, CH,
                                       nullptr, comb, 192, 128);  // comb[:,:,128:192]

  // ---- fusion ----  (bias bfu cancels under BN)
  k_gemm_wmma<<<128, 256, 0, stream>>>(wfuswz, nullptr, comb, 192,
                                       fuslin, nullptr, 0, 0);
  k_chan_stats<<<CH, 256, 0, stream>>>(fuslin, NP, mfu, sfu);
  k_bn_act<<<(BB * CH * NP) / 256, 256, 0, stream>>>(fuslin, mfu, sfu, gfu, befu, 0.0f,
                                                     NP, outp, nullptr, 0, 0);

  // second output: coords passthrough
  hipMemcpyAsync(outp + (size_t)BB * CH * NP, coords,
                 (size_t)BB * 3 * NP * sizeof(float),
                 hipMemcpyDeviceToDevice, stream);
}
